// SparseTopKMoE_34583076667493
// MI455X (gfx1250) — matrix operand order repaired and checked by reading, not compiled, not run on hardware
//
#include <hip/hip_runtime.h>
#include <hip/hip_bf16.h>
#include <math.h>

// ---------------- problem constants ----------------
#define HH     1024          // hidden
#define IS     2048          // intermediate
#define NE     8             // routed experts
#define NX     9             // routed + shared
#define TOPK   2
#define TTOK   4096          // B*S
#define KTOT   (NX * IS)     // 18432 : concatenated K for the down GEMM

// d_out layout (fp32): routed [T,H] | aux | z | logits [T,E]
#define OUT_ROUTED_N   ((size_t)TTOK * HH)          // 4194304
#define OUT_AUX_OFF    OUT_ROUTED_N
#define OUT_LOGIT_OFF  (OUT_ROUTED_N + 2)

// ws layout (bytes)
#define WS_XB_OFF      ((size_t)0)                               // bf16 x : T*H*2
#define WS_ACT_OFF     (WS_XB_OFF + (size_t)TTOK * HH * 2)       // bf16 act: T*KTOT*2
#define WS_COMB_OFF    (WS_ACT_OFF + (size_t)TTOK * KTOT * 2)    // f32 combine: T*NX
#define WS_RED_OFF     (WS_COMB_OFF + (size_t)TTOK * NX * 4)     // f32 red[17]

typedef __bf16 v16bf __attribute__((ext_vector_type(16)));
typedef __bf16 v8bf  __attribute__((ext_vector_type(8)));
typedef __bf16 v2bf  __attribute__((ext_vector_type(2)));
typedef float  v8f   __attribute__((ext_vector_type(8)));
typedef int    v4i   __attribute__((ext_vector_type(4)));

// pack two fp32 into {lo,hi} bf16 pair; prefer HW cvt, else native bf16 casts
__device__ __forceinline__ unsigned int pack_bf16(float a, float b) {
#if __has_builtin(__builtin_amdgcn_cvt_pk_bf16_f32)
    auto r = __builtin_amdgcn_cvt_pk_bf16_f32(a, b);
    if constexpr (sizeof(r) == 4) return __builtin_bit_cast(unsigned int, r);
#endif
    v2bf pk;
    pk[0] = (__bf16)a;
    pk[1] = (__bf16)b;
    return __builtin_bit_cast(unsigned int, pk);
}

// fast silu: x * rcp(1+exp(-x)) using v_rcp_f32 (bf16 storage absorbs 1-ulp error)
__device__ __forceinline__ float fast_silu(float g) {
    return g * __builtin_amdgcn_rcpf(1.0f + __expf(-g));
}

// ---------------- async global->LDS 16B copy (CDNA5 ASYNCcnt path) ----------------
#if __has_builtin(__builtin_amdgcn_global_load_async_to_lds_b128)
#define HAS_ASYNC_LDS 1
#else
#define HAS_ASYNC_LDS 0
#endif

__device__ __forceinline__ void lds_cp16(unsigned short* lds_dst,
                                         const unsigned short* gsrc) {
#if HAS_ASYNC_LDS
    __builtin_amdgcn_global_load_async_to_lds_b128(
        (__attribute__((address_space(1))) v4i*)(unsigned short*)gsrc,
        (__attribute__((address_space(3))) v4i*)lds_dst, 0, 0);
#else
    *(uint4*)lds_dst = *(const uint4*)gsrc;
#endif
}

__device__ __forceinline__ void wait_async_lds() {
#if HAS_ASYNC_LDS
#if __has_builtin(__builtin_amdgcn_s_wait_asynccnt)
    __builtin_amdgcn_s_wait_asynccnt(0);
#else
    asm volatile("s_wait_asynccnt 0" ::: "memory");
#endif
#endif
}

// Build a v16bf fragment from two 16B-aligned 8-element LDS runs.
__device__ __forceinline__ v16bf frag2x8(const unsigned short* lo_p,
                                         const unsigned short* hi_p) {
    v8bf lo = *(const v8bf*)lo_p;
    v8bf hi = *(const v8bf*)hi_p;
    v16bf r;
#pragma unroll
    for (int i = 0; i < 8; ++i) { r[i] = lo[i]; r[i + 8] = hi[i]; }
    return r;
}

// ---------------- small kernels ----------------
__global__ void init_red_kernel(float* __restrict__ red) {
    if (threadIdx.x < 17) red[threadIdx.x] = 0.0f;
}

__global__ __launch_bounds__(256)
void cvt_x_kernel(const float* __restrict__ x, unsigned int* __restrict__ xb2) {
    int i = blockIdx.x * 256 + threadIdx.x;   // grid covers exactly T*H/2
    float2 v = ((const float2*)x)[i];
    xb2[i] = pack_bf16(v.x, v.y);
}

// Router: logits, softmax, top-2, combine weights, aux/z partial sums.
__global__ __launch_bounds__(256)
void router_kernel(const float* __restrict__ x, const float* __restrict__ rw,
                   float* __restrict__ logits, float* __restrict__ combine,
                   float* __restrict__ red) {
    __shared__ float sRW[HH * NE];   // 32 KiB
    for (int i = threadIdx.x; i < HH * NE; i += 256) sRW[i] = rw[i];
    __syncthreads();

    const int t = blockIdx.x * 256 + threadIdx.x;   // grid = T/256
    const float* xr = x + (size_t)t * HH;
    float acc[NE];
#pragma unroll
    for (int e = 0; e < NE; ++e) acc[e] = 0.0f;
    for (int h = 0; h < HH; ++h) {
        float xv = xr[h];
#pragma unroll
        for (int e = 0; e < NE; ++e) acc[e] += xv * sRW[h * NE + e];
    }
#pragma unroll
    for (int e = 0; e < NE; ++e) logits[(size_t)t * NE + e] = acc[e];

    float m = acc[0];
#pragma unroll
    for (int e = 1; e < NE; ++e) m = fmaxf(m, acc[e]);
    float p[NE], sum = 0.0f;
#pragma unroll
    for (int e = 0; e < NE; ++e) { p[e] = __expf(acc[e] - m); sum += p[e]; }
    float inv = 1.0f / sum;
    float lse = m + __logf(sum);

    int i1 = 0;
#pragma unroll
    for (int e = 1; e < NE; ++e) if (p[e] > p[i1]) i1 = e;   // lowest index on tie
    int i2 = (i1 == 0) ? 1 : 0;
#pragma unroll
    for (int e = 0; e < NE; ++e) if (e != i1 && p[e] > p[i2]) i2 = e;

    float w1 = p[i1] * inv, w2 = p[i2] * inv;
    float wn = 1.0f / (w1 + w2);
    float cv[NX];
#pragma unroll
    for (int j = 0; j < NX; ++j) cv[j] = 0.0f;
    cv[i1] = w1 * wn; cv[i2] = w2 * wn; cv[NE] = 1.0f;       // shared expert weight = 1
#pragma unroll
    for (int j = 0; j < NX; ++j) combine[(size_t)t * NX + j] = cv[j];

    atomicAdd(&red[i1], 1.0f);
    atomicAdd(&red[i2], 1.0f);
#pragma unroll
    for (int e = 0; e < NE; ++e) atomicAdd(&red[8 + e], p[e] * inv);
    atomicAdd(&red[16], lse * lse);
}

__global__ void finalize_kernel(const float* __restrict__ red, float* __restrict__ auxz) {
    float aux = 0.0f;
#pragma unroll
    for (int e = 0; e < NE; ++e)
        aux += (red[e] / (float)(TTOK * TOPK)) * (red[8 + e] / (float)TTOK);
    auxz[0] = aux * (float)NE;
    auxz[1] = red[16] / (float)TTOK;
}

// ---------------- GEMM1: act[t, e*IS + i] = combine[t,e] * silu(x@gw) * (x@uw) ----------------
// WG tile: M=64 tokens x N=128 cols; 8 waves as 2(M) x 4(N); wave = 32x32.
// grid: x = IS/128 = 16, y = T/64 = 64, z = expert 0..8
__global__ __launch_bounds__(256)
void moe_gateup_kernel(const unsigned short* __restrict__ xb,
                       const float* __restrict__ gate_w,
                       const float* __restrict__ up_w,
                       const float* __restrict__ sh_gate,
                       const float* __restrict__ sh_up,
                       const float* __restrict__ combine,
                       __bf16* __restrict__ act) {
    const int e  = blockIdx.z;
    const int n0 = blockIdx.x * 128;
    const int t0 = blockIdx.y * 64;
    const float* __restrict__ Wg = (e < NE) ? gate_w + (size_t)e * HH * IS : sh_gate;
    const float* __restrict__ Wu = (e < NE) ? up_w   + (size_t)e * HH * IS : sh_up;

    __shared__ unsigned short sA [64  * 40];  // [M][K], 80B pitch
    __shared__ unsigned short sBg[128 * 40];  // [N][K] (K-transposed)
    __shared__ unsigned short sBu[128 * 40];
    __shared__ float sC[64];

    const int tid = threadIdx.x;
    const int wave = tid >> 5, lane = tid & 31;
    const int wm = wave >> 2, wn = wave & 3;
    const int lane16 = lane & 15, half = lane >> 4;

    if (tid < 64) sC[tid] = combine[(size_t)(t0 + tid) * NX + e];

    // --- per-thread fixed roles (constant-offset addressing) ---
    const unsigned short* gA = xb + (size_t)(t0 + (tid >> 2)) * HH + (tid & 3) * 8;
    unsigned short* lA = &sA[(tid >> 2) * 40 + (tid & 3) * 8];
    const int nn = tid & 127;
    const int kb = (tid >> 7) * 2;
    const float* pWg = Wg + (size_t)kb * IS + n0 + nn;
    const float* pWu = Wu + (size_t)kb * IS + n0 + nn;
    unsigned int* dBg = (unsigned int*)&sBg[nn * 40 + kb];
    unsigned int* dBu = (unsigned int*)&sBu[nn * 40 + kb];

    v8f cg[2][2], cu[2][2];
#pragma unroll
    for (int a = 0; a < 2; ++a)
#pragma unroll
        for (int b = 0; b < 2; ++b)
#pragma unroll
            for (int r = 0; r < 8; ++r) { cg[a][b][r] = 0.0f; cu[a][b][r] = 0.0f; }

    for (int ks = 0; ks < HH / 32; ++ks) {
        lds_cp16(lA, gA);                     // async DMA of A tile chunk
        __builtin_prefetch(pWg + (size_t)32 * IS, 0, 1);
        __builtin_prefetch(pWu + (size_t)32 * IS, 0, 1);
#pragma unroll
        for (int it = 0; it < 8; ++it) {      // B tiles: fp32 -> bf16, K-transposed
            float g0 = pWg[(size_t)it * 4 * IS];
            float g1 = pWg[(size_t)it * 4 * IS + IS];
            float u0 = pWu[(size_t)it * 4 * IS];
            float u1 = pWu[(size_t)it * 4 * IS + IS];
            dBg[it * 2] = pack_bf16(g0, g1);
            dBu[it * 2] = pack_bf16(u0, u1);
        }
        wait_async_lds();
        __syncthreads();

        v16bf afr[2], bgfr[2], bufr[2];
#pragma unroll
        for (int mi = 0; mi < 2; ++mi) {
            int row = wm * 32 + mi * 16 + lane16;
            afr[mi] = frag2x8(&sA[row * 40 + half * 16], &sA[row * 40 + half * 16 + 8]);
        }
#pragma unroll
        for (int ni = 0; ni < 2; ++ni) {
            int n = wn * 32 + ni * 16 + lane16;
            bgfr[ni] = frag2x8(&sBg[n * 40 + half * 16], &sBg[n * 40 + half * 16 + 8]);
            bufr[ni] = frag2x8(&sBu[n * 40 + half * 16], &sBu[n * 40 + half * 16 + 8]);
        }
#pragma unroll
        for (int mi = 0; mi < 2; ++mi)
#pragma unroll
            for (int ni = 0; ni < 2; ++ni) {
                cg[mi][ni] = __builtin_amdgcn_wmma_f32_16x16x32_bf16(
                    false, afr[mi], false, bgfr[ni], (short)0, cg[mi][ni], false, false);
                cu[mi][ni] = __builtin_amdgcn_wmma_f32_16x16x32_bf16(
                    false, afr[mi], false, bufr[ni], (short)0, cu[mi][ni], false, false);
            }
        __syncthreads();

        gA  += 32;
        pWg += (size_t)32 * IS;
        pWu += (size_t)32 * IS;
    }

#pragma unroll
    for (int mi = 0; mi < 2; ++mi) {
        const int rbase = wm * 32 + mi * 16 + half * 8;
        __bf16* pact = act + (size_t)(t0 + rbase) * KTOT + (size_t)e * IS
                           + n0 + wn * 32 + lane16;
#pragma unroll
        for (int ni = 0; ni < 2; ++ni)
#pragma unroll
            for (int r = 0; r < 8; ++r) {
                float g = cg[mi][ni][r], u = cu[mi][ni][r];
                float v = fast_silu(g) * u * sC[rbase + r];
                pact[(size_t)r * KTOT + ni * 16] = (__bf16)v;
            }
    }
}

// ---------------- GEMM2: out[T,H] = act[T,KTOT] x downcat[KTOT,H] ----------------
// WG tile: M=64 x N=128; 8 waves 2x4; wave 32x32. grid: x = H/128 = 8, y = T/64 = 64.
__global__ __launch_bounds__(256)
void moe_down_kernel(const unsigned short* __restrict__ act,
                     const float* __restrict__ down_w,
                     const float* __restrict__ sh_down,
                     float* __restrict__ out) {
    const int n0 = blockIdx.x * 128;
    const int t0 = blockIdx.y * 64;

    __shared__ unsigned short sA[64  * 40];
    __shared__ unsigned short sB[128 * 40];

    const int tid = threadIdx.x;
    const int wave = tid >> 5, lane = tid & 31;
    const int wm = wave >> 2, wn = wave & 3;
    const int lane16 = lane & 15, half = lane >> 4;

    const unsigned short* gA = act + (size_t)(t0 + (tid >> 2)) * KTOT + (tid & 3) * 8;
    unsigned short* lA = &sA[(tid >> 2) * 40 + (tid & 3) * 8];
    const int nn = tid & 127;
    const int kb = (tid >> 7) * 2;
    unsigned int* dB = (unsigned int*)&sB[nn * 40 + kb];

    v8f c[2][2];
#pragma unroll
    for (int a = 0; a < 2; ++a)
#pragma unroll
        for (int b = 0; b < 2; ++b)
#pragma unroll
            for (int r = 0; r < 8; ++r) c[a][b][r] = 0.0f;

    for (int ks = 0; ks < KTOT / 32; ++ks) {     // 576 k-steps; each stays inside one expert
        const int k0 = ks * 32;
        const int e  = k0 >> 11;                 // k0 / IS
        const int i0 = k0 & (IS - 1);
        const float* __restrict__ W =
            ((e < NE) ? down_w + (size_t)e * IS * HH : sh_down)
            + (size_t)(i0 + kb) * HH + n0 + nn;

        lds_cp16(lA, gA);                        // async DMA of act tile chunk
        __builtin_prefetch(W + (size_t)32 * HH, 0, 1);
#pragma unroll
        for (int it = 0; it < 8; ++it) {         // B tile: fp32 -> bf16, K-transposed
            float a0 = W[(size_t)it * 4 * HH];
            float a1 = W[(size_t)it * 4 * HH + HH];
            dB[it * 2] = pack_bf16(a0, a1);
        }
        wait_async_lds();
        __syncthreads();

        v16bf afr[2], bfr[2];
#pragma unroll
        for (int mi = 0; mi < 2; ++mi) {
            int row = wm * 32 + mi * 16 + lane16;
            afr[mi] = frag2x8(&sA[row * 40 + half * 16], &sA[row * 40 + half * 16 + 8]);
        }
#pragma unroll
        for (int ni = 0; ni < 2; ++ni) {
            int n = wn * 32 + ni * 16 + lane16;
            bfr[ni] = frag2x8(&sB[n * 40 + half * 16], &sB[n * 40 + half * 16 + 8]);
        }
#pragma unroll
        for (int mi = 0; mi < 2; ++mi)
#pragma unroll
            for (int ni = 0; ni < 2; ++ni)
                c[mi][ni] = __builtin_amdgcn_wmma_f32_16x16x32_bf16(
                    false, afr[mi], false, bfr[ni], (short)0, c[mi][ni], false, false);
        __syncthreads();

        gA += 32;
    }

#pragma unroll
    for (int mi = 0; mi < 2; ++mi) {
        const int rbase = wm * 32 + mi * 16 + half * 8;
        float* pout = out + (size_t)(t0 + rbase) * HH + n0 + wn * 32 + lane16;
#pragma unroll
        for (int ni = 0; ni < 2; ++ni)
#pragma unroll
            for (int r = 0; r < 8; ++r)
                pout[(size_t)r * HH + ni * 16] = c[mi][ni][r];
    }
}

// ---------------- launch ----------------
extern "C" void kernel_launch(void* const* d_in, const int* in_sizes, int n_in,
                              void* d_out, int out_size, void* d_ws, size_t ws_size,
                              hipStream_t stream) {
    const float* x        = (const float*)d_in[0];
    const float* rw       = (const float*)d_in[1];
    const float* gate_w   = (const float*)d_in[2];
    const float* up_w     = (const float*)d_in[3];
    const float* down_w   = (const float*)d_in[4];
    const float* sh_gate  = (const float*)d_in[5];
    const float* sh_up    = (const float*)d_in[6];
    const float* sh_down  = (const float*)d_in[7];

    float* out = (float*)d_out;
    char*  ws  = (char*)d_ws;
    unsigned short* xb      = (unsigned short*)(ws + WS_XB_OFF);
    unsigned short* act     = (unsigned short*)(ws + WS_ACT_OFF);
    float*          combine = (float*)(ws + WS_COMB_OFF);
    float*          red     = (float*)(ws + WS_RED_OFF);

    init_red_kernel<<<1, 32, 0, stream>>>(red);
    cvt_x_kernel<<<(TTOK * HH / 2) / 256, 256, 0, stream>>>(x, (unsigned int*)xb);
    router_kernel<<<TTOK / 256, 256, 0, stream>>>(x, rw, out + OUT_LOGIT_OFF, combine, red);
    moe_gateup_kernel<<<dim3(IS / 128, TTOK / 64, NX), 256, 0, stream>>>(
        xb, gate_w, up_w, sh_gate, sh_up, combine, (__bf16*)act);
    moe_down_kernel<<<dim3(HH / 128, TTOK / 64), 256, 0, stream>>>(act, down_w, sh_down, out);
    finalize_kernel<<<1, 1, 0, stream>>>(red, out + OUT_AUX_OFF);
}
